// Linear3Bit_90348932039240
// MI455X (gfx1250) — matrix operand order repaired and checked by reading, not compiled, not run on hardware
//
#include <hip/hip_runtime.h>
#include <hip/hip_bf16.h>

typedef __attribute__((ext_vector_type(16))) _Float16 v16h;
typedef __attribute__((ext_vector_type(8)))  _Float16 v8h;
typedef __attribute__((ext_vector_type(8)))  float    v8f;
typedef __attribute__((ext_vector_type(4)))  float    v4f;

#define OUT_F   11008
#define IN_F    4096
#define KTILES  (IN_F / 32)      // 128 k-steps of 32
#define NB      128              // N per workgroup (8 waves x 16)
#define MB      128              // M per workgroup (8 WMMA M-tiles per wave)
#define MT      8                // M-tiles per wave

// Dequantize one 3-byte triple (stored as 3 int32s, one byte each) -> 8 f16 values.
// Bit-exact with the reference unpack: q in [0,7], w = q*(2*norm/7) - norm.
__device__ __forceinline__ v8h dq8(int b0, int b1, int b2, float s, float o) {
    v8h r;
    r[0] = (_Float16)fmaf((float)( b0        & 7), s, o);
    r[1] = (_Float16)fmaf((float)((b0 >> 3)  & 7), s, o);
    r[2] = (_Float16)fmaf((float)(((b0 >> 6) & 3) | ((b1 & 1) << 2)), s, o);
    r[3] = (_Float16)fmaf((float)((b1 >> 1)  & 7), s, o);
    r[4] = (_Float16)fmaf((float)((b1 >> 4)  & 7), s, o);
    r[5] = (_Float16)fmaf((float)(((b1 >> 7) & 1) | ((b2 & 3) << 1)), s, o);
    r[6] = (_Float16)fmaf((float)((b2 >> 2)  & 7), s, o);
    r[7] = (_Float16)fmaf((float)((b2 >> 5)  & 7), s, o);
    return r;
}

// fp32 x-tile fragment (16 floats) -> two v8h for LDS staging
__device__ __forceinline__ void cvt16(v4f f0, v4f f1, v4f f2, v4f f3,
                                      v8h& h0, v8h& h1) {
#pragma unroll
    for (int j = 0; j < 4; ++j) {
        h0[j]     = (_Float16)f0[j];
        h0[j + 4] = (_Float16)f1[j];
        h1[j]     = (_Float16)f2[j];
        h1[j + 4] = (_Float16)f3[j];
    }
}

__global__ __launch_bounds__(256)
void Linear3Bit_90348932039240_kernel(const float* __restrict__ x,
                                      const int* __restrict__ wq3,
                                      const _Float16* __restrict__ wnorm,
                                      const float* __restrict__ bias,
                                      float* __restrict__ out) {
    // Ping-pong LDS: 2 x (128 rows x 32 k) f16 = 16 KB (of 320 KB/WGP)
    __shared__ _Float16 sx[2][MB * 32];

    const int tid    = threadIdx.x;
    const int lane   = tid & 31;
    const int wave   = tid >> 5;
    const int laneN  = lane & 15;    // column within 16-wide N tile
    const int laneHi = lane >> 4;    // which K-half this lane owns

    const int nbase = blockIdx.x * NB + wave * 16;
    const int o     = nbase + laneN;               // this lane's output feature
    const int mbase = blockIdx.y * MB;

    // Weight stream: 128 groups per out-row, 12 ints per group;
    // this lane's half of the group = 6 consecutive ints (two 3-byte triples).
    const int*       wq = wq3   + (size_t)o * (KTILES * 12) + laneHi * 6;
    const _Float16*  np = wnorm + (size_t)o * KTILES;

    // x staging role: thread handles row r, 16 consecutive k (two halves of 32)
    const int r = tid >> 1;              // 0..127
    const int c = (tid & 1) << 4;        // 0 or 16
    const float* xrow = x + (size_t)(mbase + r) * IN_F + c;

    v8f acc[MT] = {};   // 8 x (16x16 f32 tile) accumulators

    // ---------------- Prologue: stage k-step 0, load its weights ----------------
    {
        const v4f* xp = (const v4f*)xrow;
        v4f f0 = xp[0], f1 = xp[1], f2 = xp[2], f3 = xp[3];
        v8h h0, h1;
        cvt16(f0, f1, f2, f3, h0, h1);
        *(v8h*)&sx[0][r * 32 + c]     = h0;
        *(v8h*)&sx[0][r * 32 + c + 8] = h1;
    }
    int2  c0 = *(const int2*)(wq);
    int2  c1 = *(const int2*)(wq + 2);
    int2  c2 = *(const int2*)(wq + 4);
    float cn = (float)np[0];

    // ---------------- Main pipelined loop: one barrier per k-step ----------------
    for (int kt = 0; kt < KTILES; ++kt) {
        const int p = kt & 1;
        __syncthreads();                 // sx[p] staged; sx[1-p] free for refill

        // Issue next k-step's global loads early (overlap with dequant + WMMA)
        const bool more = (kt + 1) < KTILES;
        v4f g0, g1, g2, g3;
        int2 n0, n1, n2;
        float nn;
        if (more) {
            const v4f* xp = (const v4f*)(xrow + (kt + 1) * 32);
            g0 = xp[0]; g1 = xp[1]; g2 = xp[2]; g3 = xp[3];
            const int* wqn = wq + 12;
            n0 = *(const int2*)(wqn);
            n1 = *(const int2*)(wqn + 2);
            n2 = *(const int2*)(wqn + 4);
            nn = (float)np[kt + 1];
            // warm L2 for the dominant weight stream, ~8 k-steps (384 B) ahead
            __builtin_prefetch(wq + 12 * 8, 0, 0);
        }

        // ---- B fragment: dequantize this lane's 16 K-values of its out-row ----
        float scl = cn * (2.0f / 7.0f);
        float off = -cn;
        v8h b_lo = dq8(c0.x, c0.y, c1.x, scl, off);
        v8h b_hi = dq8(c1.y, c2.x, c2.y, scl, off);
        v16h bfrag = __builtin_shufflevector(b_lo, b_hi,
                        0,1,2,3,4,5,6,7,8,9,10,11,12,13,14,15);

        // ---- 8 WMMAs: A fragments from sx[p] in contiguous k order (16*laneHi + s), same as B ----
        const int kofs = laneHi * 16;    // halves: 0..15 / 16..31
#pragma unroll
        for (int t = 0; t < MT; ++t) {
            const int row = t * 16 + laneN;
            v8h alo = *(const v8h*)&sx[p][row * 32 + kofs];       // K x..x+7
            v8h ahi = *(const v8h*)&sx[p][row * 32 + kofs + 8];   // K x+8..x+15
            v16h afrag = __builtin_shufflevector(alo, ahi,
                            0,1,2,3,4,5,6,7,8,9,10,11,12,13,14,15);
            acc[t] = __builtin_amdgcn_wmma_f32_16x16x32_f16(
                         /*neg_a=*/false, afrag, /*neg_b=*/false, bfrag,
                         /*c_mod=*/(short)0, acc[t],
                         /*reuse_a=*/false, /*reuse_b=*/false);
        }

        // ---- Stage next tile into sx[1-p]; next barrier publishes it ----
        if (more) {
            v8h h0, h1;
            cvt16(g0, g1, g2, g3, h0, h1);
            *(v8h*)&sx[1 - p][r * 32 + c]     = h0;
            *(v8h*)&sx[1 - p][r * 32 + c + 8] = h1;
            c0 = n0; c1 = n1; c2 = n2; cn = nn;
        }
        wq += 12;
    }

    // ---- Epilogue: bias add + store per documented C/D layout ----
    const float bv = bias[o];
#pragma unroll
    for (int t = 0; t < MT; ++t) {
#pragma unroll
        for (int j = 0; j < 8; ++j) {
            const int m = mbase + t * 16 + (laneHi ? j + 8 : j);
            out[(size_t)m * OUT_F + o] = acc[t][j] + bv;
        }
    }
}

extern "C" void kernel_launch(void* const* d_in, const int* in_sizes, int n_in,
                              void* d_out, int out_size, void* d_ws, size_t ws_size,
                              hipStream_t stream) {
    const float*    x     = (const float*)d_in[0];
    const int*      wq3   = (const int*)d_in[1];
    const _Float16* wnorm = (const _Float16*)d_in[2];
    const float*    bias  = (const float*)d_in[3];
    float*          out   = (float*)d_out;

    dim3 grid(OUT_F / NB, 512 / MB);   // (86, 4)
    dim3 block(256);                   // 8 wave32
    Linear3Bit_90348932039240_kernel<<<grid, block, 0, stream>>>(x, wq3, wnorm, bias, out);
}
